// Attention_76639396430542
// MI455X (gfx1250) — compile-verified, matrix operand order checked by reading, not run on hardware
//
#include <hip/hip_runtime.h>

#define B_  2
#define S_  2048
#define H_  16
#define HD_ 64
#define D_  1024

typedef __attribute__((ext_vector_type(16))) __bf16 v16bf;
typedef __attribute__((ext_vector_type(8)))  float  v8f;
typedef int gvec4i __attribute__((vector_size(16)));   // matches builtin's V4i

#define AS1 __attribute__((address_space(1)))
#define AS3 __attribute__((address_space(3)))

#if __has_builtin(__builtin_amdgcn_global_load_async_to_lds_b128) && \
    __has_builtin(__builtin_amdgcn_s_wait_asynccnt)
#define HAVE_ASYNC 1
// src: global (AS1) int4*, dst: LDS (AS3) int4*, imm byte offset, imm cpol
#define ASYNC_LDS_B128(src, dst, off)                                          \
  __builtin_amdgcn_global_load_async_to_lds_b128(                              \
      (AS1 gvec4i*)(AS1 void*)(void*)(src),                                    \
      (AS3 gvec4i*)(AS3 void*)(dst), (off), 0)
#else
#define HAVE_ASYNC 0
#endif

union FragU { unsigned u[8]; uint4 q[2]; v16bf v; };

static __device__ __forceinline__ unsigned rne_bits(float f) {
  unsigned u = __builtin_bit_cast(unsigned, f);
  return u + 0x7FFFu + ((u >> 16) & 1u);     // round-to-nearest-even bias
}
static __device__ __forceinline__ unsigned short f2bf(float f) {
  return (unsigned short)(rne_bits(f) >> 16);
}
// two f32 -> one dword holding 2 bf16, single v_perm_b32 for the pack
static __device__ __forceinline__ unsigned pack2bf(float f0, float f1) {
  return __builtin_amdgcn_perm(rne_bits(f1), rne_bits(f0), 0x07060302u);
}

// 16-element bf16 fragment: lane's K values are runs {base..base+7} and
// {base+16..base+23}, base = (lane<16 ? 0 : 8)  -> two b128 loads.
static __device__ __forceinline__ v16bf frag_bf16(const unsigned short* row, int base) {
  FragU f;
  f.q[0] = *(const uint4*)(row + base);
  f.q[1] = *(const uint4*)(row + base + 16);
  return f.v;
}

static __device__ __forceinline__ v8f wmma_bf16(v16bf a, v16bf b, v8f c) {
  return __builtin_amdgcn_wmma_f32_16x16x32_bf16(false, a, false, b, (short)0, c, false, false);
}

// ---------------------------------------------------------------------------
// Kernel 0: fp32 -> bf16 bulk conversion (memory bound, perm-packed b128 I/O)
// ---------------------------------------------------------------------------
__global__ __launch_bounds__(256) void cvt_kernel(const float* __restrict__ in,
                                                  unsigned short* __restrict__ out,
                                                  int n8) {
  int i = blockIdx.x * 256 + threadIdx.x;
  if (i >= n8) return;
  float4 a = ((const float4*)in)[2 * i];
  float4 b = ((const float4*)in)[2 * i + 1];
  uint4 r;
  r.x = pack2bf(a.x, a.y);
  r.y = pack2bf(a.z, a.w);
  r.z = pack2bf(b.x, b.y);
  r.w = pack2bf(b.z, b.w);
  ((uint4*)out)[i] = r;
}

// ---------------------------------------------------------------------------
// Kernel 1: per-head projections, all-bf16 WMMA.
//   Q,K out: [B*H, S, 64] token-major (Q folded with 1/sqrt(hd))
//   V  out:  [B*H, 64, S] dim-major (pre-transposed, packed b128 stores)
// ---------------------------------------------------------------------------
__global__ __launch_bounds__(128) void proj_kernel(
    const unsigned short* __restrict__ Xq, const unsigned short* __restrict__ Xk,
    const unsigned short* __restrict__ Xv,
    const unsigned short* __restrict__ Wqb, const unsigned short* __restrict__ Wkb,
    const unsigned short* __restrict__ Wvb,
    const float* __restrict__ bq, const float* __restrict__ bk,
    const float* __restrict__ bv,
    unsigned short* __restrict__ Qp, unsigned short* __restrict__ Kp,
    unsigned short* __restrict__ Vt)
{
  const int lane  = threadIdx.x & 31;
  const int wv    = threadIdx.x >> 5;
  const int n15   = lane & 15;
  const int kbase = (lane < 16) ? 0 : 8;
  const int mofs  = (lane < 16) ? 0 : 8;

  int t   = blockIdx.x * 4 + wv;     // 0 .. 3*4096-1
  int sel = t >> 12;                 // 0=Q 1=K 2=V
  int rt  = t & 4095;                // 16-row tile over B*H*S rows

  const unsigned short* X; const unsigned short* W; const float* Bv;
  if (sel == 0)      { X = Xq; W = Wqb; Bv = bq; }
  else if (sel == 1) { X = Xk; W = Wkb; Bv = bk; }
  else               { X = Xv; W = Wvb; Bv = bv; }

  // lane's A row: global (b,h,s) row index (row = bh*S + s)
  int row = rt * 16 + n15;
  int b   = row >> 15;
  int hs  = row & 32767;
  int h   = hs >> 11;
  int s   = hs & 2047;
  const unsigned short* xr = X + (((size_t)b * S_ + s) * H_ + h) * HD_;
  v16bf a0 = frag_bf16(xr, kbase);
  v16bf a1 = frag_bf16(xr + 32, kbase);

  v8f c[4];
  #pragma unroll
  for (int nt = 0; nt < 4; ++nt) {
    const unsigned short* wr = W + (size_t)(nt * 16 + n15) * HD_;  // x @ W^T
    v8f acc = {};
    acc = wmma_bf16(a0, frag_bf16(wr, kbase), acc);
    acc = wmma_bf16(a1, frag_bf16(wr + 32, kbase), acc);
    c[nt] = acc;
  }

  if (sel < 2) {
    unsigned short* Op = (sel == 0) ? Qp : Kp;
    float scale = (sel == 0) ? 0.125f : 1.0f;   // fold 1/sqrt(64) into Q
    #pragma unroll
    for (int nt = 0; nt < 4; ++nt) {
      float bias = Bv[nt * 16 + n15];
      #pragma unroll
      for (int r = 0; r < 8; ++r) {
        int mrow = rt * 16 + r + mofs;
        Op[(size_t)mrow * HD_ + nt * 16 + n15] = f2bf((c[nt][r] + bias) * scale);
      }
    }
  } else {
    // transposed V: rows of C-layout are 8 consecutive tokens -> packed b128
    int bh = row >> 11;                 // same for whole tile
    int s0 = (rt * 16) & 2047;          // first token of tile
    #pragma unroll
    for (int nt = 0; nt < 4; ++nt) {
      float bias = Bv[nt * 16 + n15];
      uint4 pk;
      pk.x = pack2bf(c[nt][0] + bias, c[nt][1] + bias);
      pk.y = pack2bf(c[nt][2] + bias, c[nt][3] + bias);
      pk.z = pack2bf(c[nt][4] + bias, c[nt][5] + bias);
      pk.w = pack2bf(c[nt][6] + bias, c[nt][7] + bias);
      unsigned short* vp = Vt + (size_t)bh * HD_ * S_ +
                           (size_t)(nt * 16 + n15) * S_ + s0 + mofs;
      *(uint4*)vp = pk;
    }
  }
}

// ---------------------------------------------------------------------------
// Kernel 2: flash attention, double-buffered async-to-LDS pipeline.
// Block = 4 waves = 64 queries of one (b,h). Per 64-key tile:
//   8 WMMA scores, online softmax (max via shuffles), 8 WMMA P@V,
//   2 WMMA P@ones for the softmax denominator (no sum reductions).
// ---------------------------------------------------------------------------
__global__ __launch_bounds__(128) void flash_kernel(
    const unsigned short* __restrict__ Qp,
    const unsigned short* __restrict__ Kp,
    const unsigned short* __restrict__ Vt,   // [B*H, 64, S] dim-major
    unsigned short* __restrict__ Ob)         // [B, S, D] bf16
{
  __shared__ __align__(16) unsigned short sK[2][64 * 64];  // [token][dim]
  __shared__ __align__(16) unsigned short sV[2][64 * 64];  // [dim][token]
  __shared__ __align__(16) unsigned short sP[4][16 * 64];  // per-wave P

  const int lane  = threadIdx.x & 31;
  const int wv    = threadIdx.x >> 5;
  const int n15   = lane & 15;
  const int kbase = (lane < 16) ? 0 : 8;
  const int mofs  = (lane < 16) ? 0 : 8;

  const int QT = S_ / 64;
  int qt = blockIdx.x % QT;
  int bh = blockIdx.x / QT;
  int b  = bh >> 4;
  int h  = bh & 15;
  int q0 = qt * 64 + wv * 16;

  const unsigned short* Qrow = Qp + ((size_t)bh * S_ + q0 + n15) * HD_;
  v16bf aq0 = frag_bf16(Qrow, kbase);
  v16bf aq1 = frag_bf16(Qrow + 32, kbase);

  // constant "ones column" B fragment: dim n15==0 is 1.0, rest 0 -> l = P @ 1
  v16bf b_one;
  {
    FragU fo;
    unsigned fill = (n15 == 0) ? 0x3F803F80u : 0u;
    #pragma unroll
    for (int i = 0; i < 8; ++i) fo.u[i] = fill;
    b_one = fo.v;
  }

  v8f acc[4] = {};
  v8f accl   = {};                 // per-row softmax denominators (col 0)
  float mrow[8];
  #pragma unroll
  for (int r = 0; r < 8; ++r) mrow[r] = -1e30f;

  // staging: thread -> (row, 32-elem half); 4x b128 per thread per tile
  const int srow  = threadIdx.x >> 1;
  const int shalf = (threadIdx.x & 1) * 32;

  const unsigned short* Kg = Kp + (size_t)bh * S_ * HD_;
  const unsigned short* Vg = Vt + (size_t)bh * HD_ * S_;

  auto stage = [&](int buf, int kb) {
    const unsigned short* kgp = Kg + (size_t)(kb + srow) * HD_ + shalf;
    const unsigned short* vgp = Vg + (size_t)srow * S_ + kb + shalf;
    unsigned short* kls = sK[buf] + srow * 64 + shalf;
    unsigned short* vls = sV[buf] + srow * 64 + shalf;
#if HAVE_ASYNC
    ASYNC_LDS_B128(kgp, kls, 0);
    ASYNC_LDS_B128(kgp, kls, 16);
    ASYNC_LDS_B128(kgp, kls, 32);
    ASYNC_LDS_B128(kgp, kls, 48);
    ASYNC_LDS_B128(vgp, vls, 0);
    ASYNC_LDS_B128(vgp, vls, 16);
    ASYNC_LDS_B128(vgp, vls, 32);
    ASYNC_LDS_B128(vgp, vls, 48);
#else
    const uint4* kg4 = (const uint4*)kgp;
    const uint4* vg4 = (const uint4*)vgp;
    uint4* kl4 = (uint4*)kls;
    uint4* vl4 = (uint4*)vls;
    kl4[0] = kg4[0]; kl4[1] = kg4[1]; kl4[2] = kg4[2]; kl4[3] = kg4[3];
    vl4[0] = vg4[0]; vl4[1] = vg4[1]; vl4[2] = vg4[2]; vl4[3] = vg4[3];
#endif
  };

  stage(0, 0);                     // prologue: tile 0 in flight

  for (int j = 0; j < QT; ++j) {   // QT == S/64 tiles
    const int buf = j & 1;
#if HAVE_ASYNC
    __builtin_amdgcn_s_wait_asynccnt(0);   // my async writes to LDS done
#endif
    __syncthreads();                        // everyone's writes/reads ordered
    if (j + 1 < QT) stage(1 - buf, (j + 1) * 64);   // overlap with compute

    // scores: S = Q (16x64) @ K^T (64 keys), 4 key column tiles
    v8f c[4] = {};
    #pragma unroll
    for (int nt = 0; nt < 4; ++nt) {
      const unsigned short* kr = sK[buf] + (size_t)(nt * 16 + n15) * 64;
      c[nt] = wmma_bf16(aq0, frag_bf16(kr, kbase), c[nt]);
      c[nt] = wmma_bf16(aq1, frag_bf16(kr + 32, kbase), c[nt]);
    }

    // online softmax over 64 keys: max via 16-lane-group shuffles
    #pragma unroll
    for (int r = 0; r < 8; ++r) {
      float s0 = c[0][r], s1 = c[1][r], s2 = c[2][r], s3 = c[3][r];
      float mx = fmaxf(fmaxf(s0, s1), fmaxf(s2, s3));
      mx = fmaxf(mx, __shfl_xor(mx, 1, 16));
      mx = fmaxf(mx, __shfl_xor(mx, 2, 16));
      mx = fmaxf(mx, __shfl_xor(mx, 4, 16));
      mx = fmaxf(mx, __shfl_xor(mx, 8, 16));
      float mnew = fmaxf(mrow[r], mx);
      float corr = exp2f((mrow[r] - mnew) * 1.44269504f);
      float p0 = exp2f((s0 - mnew) * 1.44269504f);
      float p1 = exp2f((s1 - mnew) * 1.44269504f);
      float p2 = exp2f((s2 - mnew) * 1.44269504f);
      float p3 = exp2f((s3 - mnew) * 1.44269504f);
      mrow[r] = mnew;
      #pragma unroll
      for (int nt = 0; nt < 4; ++nt) acc[nt][r] *= corr;
      accl[r] *= corr;
      unsigned short* pr = &sP[wv][(r + mofs) * 64];
      pr[n15]      = f2bf(p0);
      pr[16 + n15] = f2bf(p1);
      pr[32 + n15] = f2bf(p2);
      pr[48 + n15] = f2bf(p3);
    }

    // P (16x64) @ V (64x64) + P @ ones; same-wave LDS ops are in order
    v16bf ap0 = frag_bf16(&sP[wv][(size_t)n15 * 64], kbase);
    v16bf ap1 = frag_bf16(&sP[wv][(size_t)n15 * 64 + 32], kbase);
    #pragma unroll
    for (int nt = 0; nt < 4; ++nt) {
      const unsigned short* vr = sV[buf] + (size_t)(nt * 16 + n15) * 64;
      acc[nt] = wmma_bf16(ap0, frag_bf16(vr, kbase), acc[nt]);
      acc[nt] = wmma_bf16(ap1, frag_bf16(vr + 32, kbase), acc[nt]);
    }
    accl = wmma_bf16(ap0, b_one, accl);
    accl = wmma_bf16(ap1, b_one, accl);
  }

  // epilogue: broadcast l from col 0 of each 16-lane group, normalize, store
  #pragma unroll
  for (int r = 0; r < 8; ++r) {
    float l   = __shfl(accl[r], lane & 16, 32);
    float inv = 1.0f / l;
    int m = q0 + r + mofs;
    #pragma unroll
    for (int nt = 0; nt < 4; ++nt) {
      Ob[((size_t)b * S_ + m) * D_ + h * HD_ + nt * 16 + n15] = f2bf(acc[nt][r] * inv);
    }
  }
}

// ---------------------------------------------------------------------------
// Kernel 3: final FC, out = O @ Wfc^T + bfc. 32 rows x 64 cols per wave so
// each W fragment feeds two WMMAs.
// ---------------------------------------------------------------------------
__global__ __launch_bounds__(128) void fc_kernel(
    const unsigned short* __restrict__ Ob,    // [B*S, D] bf16
    const unsigned short* __restrict__ Wfcb,  // [D, D] bf16
    const float* __restrict__ bfc,
    float* __restrict__ out)
{
  const int lane  = threadIdx.x & 31;
  const int wv    = threadIdx.x >> 5;
  const int n15   = lane & 15;
  const int kbase = (lane < 16) ? 0 : 8;
  const int mofs  = (lane < 16) ? 0 : 8;

  int m0 = blockIdx.x * 128 + wv * 32;
  int c0 = blockIdx.y * 64;

  v8f acc[2][4] = {};
  const unsigned short* arow0 = Ob + (size_t)(m0 + n15) * D_;
  const unsigned short* arow1 = Ob + (size_t)(m0 + 16 + n15) * D_;

  for (int kc = 0; kc < D_; kc += 32) {
    v16bf a0 = frag_bf16(arow0 + kc, kbase);
    v16bf a1 = frag_bf16(arow1 + kc, kbase);
    #pragma unroll
    for (int nt = 0; nt < 4; ++nt) {
      const unsigned short* wr = Wfcb + (size_t)(c0 + nt * 16 + n15) * D_ + kc;
      v16bf w = frag_bf16(wr, kbase);
      acc[0][nt] = wmma_bf16(a0, w, acc[0][nt]);
      acc[1][nt] = wmma_bf16(a1, w, acc[1][nt]);
    }
  }

  #pragma unroll
  for (int mt = 0; mt < 2; ++mt) {
    #pragma unroll
    for (int nt = 0; nt < 4; ++nt) {
      float bias = bfc[c0 + nt * 16 + n15];
      #pragma unroll
      for (int r = 0; r < 8; ++r) {
        int m = m0 + mt * 16 + r + mofs;
        out[(size_t)m * D_ + c0 + nt * 16 + n15] = acc[mt][nt][r] + bias;
      }
    }
  }
}

// ---------------------------------------------------------------------------
extern "C" void kernel_launch(void* const* d_in, const int* in_sizes, int n_in,
                              void* d_out, int out_size, void* d_ws, size_t ws_size,
                              hipStream_t stream) {
  (void)in_sizes; (void)n_in; (void)out_size; (void)ws_size;
  const float* q   = (const float*)d_in[0];
  const float* k   = (const float*)d_in[1];
  const float* v   = (const float*)d_in[2];
  const float* Wq  = (const float*)d_in[3];
  const float* bq  = (const float*)d_in[4];
  const float* Wk  = (const float*)d_in[5];
  const float* bk  = (const float*)d_in[6];
  const float* Wv  = (const float*)d_in[7];
  const float* bv  = (const float*)d_in[8];
  const float* Wfc = (const float*)d_in[9];
  const float* bfc = (const float*)d_in[10];
  float* out = (float*)d_out;

  const size_t nX  = (size_t)B_ * S_ * D_;   // 4M elems
  const size_t nW  = (size_t)HD_ * HD_;      // 4096
  const size_t nFC = (size_t)D_ * D_;        // 1M

  unsigned short* p   = (unsigned short*)d_ws;
  unsigned short* Xq  = p; p += nX;
  unsigned short* Xk  = p; p += nX;
  unsigned short* Xv  = p; p += nX;
  unsigned short* Wqb = p; p += nW;
  unsigned short* Wkb = p; p += nW;
  unsigned short* Wvb = p; p += nW;
  unsigned short* Wfb = p; p += nFC;
  unsigned short* Qp  = p; p += nX;
  unsigned short* Kp  = p; p += nX;
  unsigned short* Vt  = p; p += nX;          // [B*H, 64, S]
  unsigned short* Ob  = p; p += nX;

  // bulk fp32 -> bf16 conversions
  cvt_kernel<<<(int)(nX / 8 / 256), 256, 0, stream>>>(q, Xq, (int)(nX / 8));
  cvt_kernel<<<(int)(nX / 8 / 256), 256, 0, stream>>>(k, Xk, (int)(nX / 8));
  cvt_kernel<<<(int)(nX / 8 / 256), 256, 0, stream>>>(v, Xv, (int)(nX / 8));
  cvt_kernel<<<2, 256, 0, stream>>>(Wq, Wqb, (int)(nW / 8));
  cvt_kernel<<<2, 256, 0, stream>>>(Wk, Wkb, (int)(nW / 8));
  cvt_kernel<<<2, 256, 0, stream>>>(Wv, Wvb, (int)(nW / 8));
  cvt_kernel<<<(int)(nFC / 8 / 256), 256, 0, stream>>>(Wfc, Wfb, (int)(nFC / 8));

  // 3 matrices * 4096 row-tiles / 4 waves per block
  proj_kernel<<<3072, 128, 0, stream>>>(Xq, Xk, Xv, Wqb, Wkb, Wvb,
                                        bq, bk, bv, Qp, Kp, Vt);
  // B*H*(S/64) blocks of 64 queries
  flash_kernel<<<B_ * H_ * (S_ / 64), 128, 0, stream>>>(Qp, Kp, Vt, Ob);
  // (4096/128) x (1024/64)
  fc_kernel<<<dim3((B_ * S_) / 128, D_ / 64), 128, 0, stream>>>(Ob, Wfb, bfc, out);
}
